// AttentionModule_4389456577454
// MI455X (gfx1250) — matrix operand order repaired and checked by reading, not compiled, not run on hardware
//
#include <hip/hip_runtime.h>
#include <stdint.h>

// ---------------- WMMA / TDM types (per CDNA5_HIP.md) ------------------------
typedef __attribute__((ext_vector_type(16))) _Float16 v16h;
typedef __attribute__((ext_vector_type(8)))  _Float16 h8;
typedef __attribute__((ext_vector_type(8)))  float    v8f;
typedef __attribute__((ext_vector_type(4)))  unsigned int ui32x4;
typedef __attribute__((ext_vector_type(8)))  int      i32x8;
typedef __attribute__((ext_vector_type(4)))  int      i32x4;

__device__ __forceinline__ float gelu_exact(float x) {
    return 0.5f * x * (1.0f + erff(x * 0.70710678118654752440f));
}

// -----------------------------------------------------------------------------
// Elementwise fp32 -> f16 convert (n multiple of 8). 8 elems / thread.
// -----------------------------------------------------------------------------
__global__ void convert_f32_to_f16(const float* __restrict__ in,
                                   _Float16* __restrict__ out, size_t n8) {
    size_t i = (size_t)blockIdx.x * blockDim.x + threadIdx.x;
    if (i >= n8) return;
    const float4* p = (const float4*)in + i * 2;
    float4 x0 = p[0], x1 = p[1];
    h8 o;
    o[0] = (_Float16)x0.x; o[1] = (_Float16)x0.y; o[2] = (_Float16)x0.z; o[3] = (_Float16)x0.w;
    o[4] = (_Float16)x1.x; o[5] = (_Float16)x1.y; o[6] = (_Float16)x1.z; o[7] = (_Float16)x1.w;
    ((h8*)out)[i] = o;
}

// -----------------------------------------------------------------------------
// Transpose + fp32 -> f16 convert (zero column padding):
//   in : (S, R, Cc) fp32 row-major -> out: (S, Cpad, R) f16
// -----------------------------------------------------------------------------
__global__ void transpose_f32_to_f16(const float* __restrict__ in,
                                     _Float16* __restrict__ out,
                                     int S, int R, int Cc, int Cpad) {
    size_t idx = (size_t)blockIdx.x * blockDim.x + threadIdx.x;
    size_t total = (size_t)S * Cpad * R;
    if (idx >= total) return;
    int r  = (int)(idx % R);
    size_t t = idx / R;
    int cp = (int)(t % Cpad);
    int s  = (int)(t / Cpad);
    float v = (cp < Cc) ? in[((size_t)s * R + r) * Cc + cp] : 0.0f;
    out[idx] = (_Float16)v;
}

// -----------------------------------------------------------------------------
// Wide WMMA GEMM with TDM-staged B in LDS.
//   C[M x Nc] = act(A @ B + bias);  A: f16 row-major, Bt: f16 [Nc x K].
// Block = 8 waves. Block owns one 64-col N strip x 8 consecutive 16-row M tiles.
// Wave 0 DMAs the B strip (64*K halves, contiguous in Bt) into LDS via
// TENSOR_LOAD_TO_LDS with 16B/row padding (bank-conflict-free ds_load_b128).
// Requires: M % 128 == 0, Nc % 64 == 0, K % 32 == 0 (K in {256,512}).
// -----------------------------------------------------------------------------
template<bool GELU, bool OUTH>
__global__ void gemm_wmma_lds(const _Float16* __restrict__ A,
                              const _Float16* __restrict__ Bt,
                              const float* __restrict__ bias, int nbias,
                              void* __restrict__ Cout,
                              int M, int K, int Nc) {
    extern __shared__ _Float16 ldsB[];     // 64 rows x (K+8) halves
    const int lane = threadIdx.x & 31;
    const int wave = threadIdx.x >> 5;
    const int gn   = Nc >> 6;
    const int bn   = blockIdx.x % gn;
    const int tm   = (blockIdx.x / gn) * 8 + wave;
    const int Kp   = K + 8;                // padded LDS row stride (halves)

    // ---- TDM: one DMA of the whole B strip into LDS (wave 0 only) ----------
    if (wave == 0) {
        const unsigned long long gaddr =
            (unsigned long long)(uintptr_t)(Bt + (size_t)bn * 64 * K);
        const unsigned int laddr = (unsigned int)(uintptr_t)ldsB; // LDS offset
        const unsigned int nq    = (unsigned int)(((unsigned)K * 64u * 2u) >> 3); // 8B units
        const unsigned int padI  = (K == 512) ? 7u : 6u; // pad every 256/128 DW
        ui32x4 g0;
        g0[0] = 1u;                                   // count=1 (valid), user D#
        g0[1] = laddr;                                // lds_addr (bytes)
        g0[2] = (unsigned int)gaddr;                  // global_addr[31:0]
        g0[3] = (unsigned int)((gaddr >> 32) & 0x1FFFFFFu) | (2u << 30); // type=2
        i32x8 g1;
        g1[0] = (int)((3u << 16) | (1u << 20) | (padI << 22) | (3u << 25));
                 // data_size=8B | pad_enable | pad_interval | pad_amount=4DW(16B)
        g1[1] = (int)((nq & 0xFFFFu) << 16);          // tensor_dim0[15:0]
        g1[2] = (int)(((nq >> 16) & 0xFFFFu) | (1u << 16)); // dim0 hi | tensor_dim1=1
        g1[3] = (int)((nq & 0xFFFFu) << 16);          // tile_dim0 = nq (1D tile)
        g1[4] = 0;                                    // tile_dim1/2 unused
        g1[5] = (int)nq;                              // tensor_dim0_stride
        g1[6] = 0; g1[7] = 0;
        i32x4 z4 = (i32x4){0, 0, 0, 0};
#if __clang_major__ >= 23
        i32x8 z8 = (i32x8){0, 0, 0, 0, 0, 0, 0, 0};
        __builtin_amdgcn_tensor_load_to_lds(g0, g1, z4, z4, z8, 0);
#else
        __builtin_amdgcn_tensor_load_to_lds(g0, g1, z4, z4, 0);
#endif
        __builtin_amdgcn_s_wait_tensorcnt(0);
    }
    __syncthreads();

    // ---- fragment addressing ------------------------------------------------
    const int l15   = lane & 15;
    const int koffA = (lane >> 4) << 3;    // A: K pairs {k..k+7, k+16..k+23}
    const int koffB = (lane >> 4) << 3;    // B: K pairs {k..k+7, k+16..k+23}
    const _Float16* arow  = A + (size_t)(tm * 16 + l15) * K + koffA;
    const _Float16* bbase = ldsB + (size_t)l15 * Kp + koffB;

    v8f c[4];
    #pragma unroll
    for (int j = 0; j < 4; ++j) c[j] = (v8f){};

    for (int k0 = 0; k0 < K; k0 += 32) {
        if (k0 + 32 < K) __builtin_prefetch(arow + k0 + 32, 0, 1);
        h8 a0 = *(const h8*)(arow + k0);
        h8 a1 = *(const h8*)(arow + k0 + 16);
        v16h a;
        #pragma unroll
        for (int t = 0; t < 8; ++t) { a[t] = a0[t]; a[8 + t] = a1[t]; }
        #pragma unroll
        for (int j = 0; j < 4; ++j) {
            const _Float16* bp = bbase + (size_t)j * 16 * Kp + k0;
            h8 b0 = *(const h8*)bp;        // ds_load_b128
            h8 b1 = *(const h8*)(bp + 16); // ds_load_b128
            v16h b;
            #pragma unroll
            for (int t = 0; t < 8; ++t) { b[t] = b0[t]; b[8 + t] = b1[t]; }
            c[j] = __builtin_amdgcn_wmma_f32_16x16x32_f16(
                false, a, false, b, (short)0, c[j], false, false);
        }
    }

    // ---- epilogue: D layout VGPR r -> row r + 8*(lane>=16), col = lane&15 ---
    const int mhalf = (lane >> 4) << 3;
    const int mbase = tm * 16 + mhalf;
    #pragma unroll
    for (int j = 0; j < 4; ++j) {
        const int ncol = bn * 64 + j * 16 + l15;
        const float bv = (ncol < nbias) ? bias[ncol] : 0.0f;
        #pragma unroll
        for (int r = 0; r < 8; ++r) {
            float x = c[j][r] + bv;
            if (GELU) x = gelu_exact(x);
            if (OUTH) ((_Float16*)Cout)[(size_t)(mbase + r) * Nc + ncol] = (_Float16)x;
            else      ((float*)Cout)   [(size_t)(mbase + r) * Nc + ncol] = x;
        }
    }
}

// -----------------------------------------------------------------------------
// Narrow WMMA GEMM (Nc = 16): wave computes 4 M-tiles, B fragment reused x4.
// -----------------------------------------------------------------------------
__global__ void gemm_wmma_n16(const _Float16* __restrict__ A,
                              const _Float16* __restrict__ Bt,
                              const float* __restrict__ bias, int nbias,
                              float* __restrict__ Cout,
                              int M, int K) {
    const int lane = threadIdx.x & 31;
    const int wave = threadIdx.x >> 5;
    const int bm   = blockIdx.x * (blockDim.x >> 5) + wave;  // 4-tile super-row
    if (bm >= (M >> 6)) return;

    const int l15   = lane & 15;
    const int koffA = (lane >> 4) << 3;
    const int koffB = (lane >> 4) << 3;
    const _Float16* arow0 = A + (size_t)(bm * 64 + l15) * K + koffA;
    const _Float16* brow  = Bt + (size_t)l15 * K + koffB;

    v8f c[4];
    #pragma unroll
    for (int i = 0; i < 4; ++i) c[i] = (v8f){};

    for (int k0 = 0; k0 < K; k0 += 32) {
        h8 b0 = *(const h8*)(brow + k0);
        h8 b1 = *(const h8*)(brow + k0 + 16);
        v16h b;
        #pragma unroll
        for (int t = 0; t < 8; ++t) { b[t] = b0[t]; b[8 + t] = b1[t]; }
        #pragma unroll
        for (int i = 0; i < 4; ++i) {
            const _Float16* ar = arow0 + (size_t)i * 16 * K + k0;
            h8 a0 = *(const h8*)ar;
            h8 a1 = *(const h8*)(ar + 16);
            v16h a;
            #pragma unroll
            for (int t = 0; t < 8; ++t) { a[t] = a0[t]; a[8 + t] = a1[t]; }
            c[i] = __builtin_amdgcn_wmma_f32_16x16x32_f16(
                false, a, false, b, (short)0, c[i], false, false);
        }
    }

    const int mhalf = (lane >> 4) << 3;
    const float bv = (l15 < nbias) ? bias[l15] : 0.0f;
    #pragma unroll
    for (int i = 0; i < 4; ++i) {
        const int mbase = bm * 64 + i * 16 + mhalf;
        #pragma unroll
        for (int r = 0; r < 8; ++r)
            Cout[(size_t)(mbase + r) * 16 + l15] = c[i][r] + bv;
    }
}

// -----------------------------------------------------------------------------
// One chord-mix step:
//   Vout[b,n,:] = Vin[b,n,:] + sum_{l<13} W[b*N+n, l] * Vin[b, cols[n,l], :]
// -----------------------------------------------------------------------------
__global__ void chord_step(const float* __restrict__ Vin,
                           float* __restrict__ Vout,
                           const float* __restrict__ W,
                           const int* __restrict__ cols) {
    const int tid = blockIdx.x * blockDim.x + threadIdx.x;
    const int e4  = tid & 63;
    const int row = tid >> 6;                 // b*N + n
    if (row >= 2 * 4096) return;
    const int n     = row & 4095;
    const int b_off = (row >> 12) << 12;      // b*N

    const float4* vin4 = (const float4*)Vin;
    float4 acc = vin4[(size_t)row * 64 + e4]; // residual
    const float* wr = W + (size_t)row * 16;
    const int*   cr = cols + (size_t)n * 13;
    #pragma unroll
    for (int l = 0; l < 13; ++l) {
        const float w = wr[l];
        const int   c = cr[l];
        float4 g = vin4[((size_t)(b_off + c)) * 64 + e4];
        acc.x += w * g.x; acc.y += w * g.y; acc.z += w * g.z; acc.w += w * g.w;
    }
    ((float4*)Vout)[(size_t)row * 64 + e4] = acc;
}

// -----------------------------------------------------------------------------
extern "C" void kernel_launch(void* const* d_in, const int* in_sizes, int n_in,
                              void* d_out, int out_size, void* d_ws, size_t ws_size,
                              hipStream_t stream) {
    const float* V    = (const float*)d_in[0];
    const float* data = (const float*)d_in[1];
    const float* gW1  = (const float*)d_in[2];
    const float* gb1  = (const float*)d_in[3];
    const float* gW2  = (const float*)d_in[4];
    const float* gb2  = (const float*)d_in[5];
    const float* fsW1 = (const float*)d_in[6];
    const float* fsb1 = (const float*)d_in[7];
    const float* fsW2 = (const float*)d_in[8];
    const float* fsb2 = (const float*)d_in[9];
    const int*   cols = (const int*)d_in[10];
    float* out = (float*)d_out;

    constexpr int Bn = 2, N = 4096, E = 256, H = 512, NW = 12, NL = 13;
    constexpr int M = Bn * N;                 // 8192 rows

    // ---- workspace layout (~26 MB) ----
    char* ws = (char*)d_ws;
    _Float16* Hbuf  = (_Float16*)ws;                              // M*H f16 = 8 MB
    float*    Vbuf  = (float*)ws;                                 // M*E f32 = 8 MB (after GEMMs)
    float*    Wall  = (float*)(ws + (size_t)8 * 1024 * 1024);     // NW*M*16 f32 = 6 MB
    _Float16* dataH = (_Float16*)(ws + (size_t)14 * 1024 * 1024); // M*E f16 = 4 MB
    _Float16* Vh    = (_Float16*)(ws + (size_t)18 * 1024 * 1024); // M*E f16 = 4 MB
    _Float16* wts   = (_Float16*)(ws + (size_t)22 * 1024 * 1024); // ~3.9 MB
    _Float16* gW1t  = wts;                                        // H*E
    _Float16* gW2t  = gW1t  + (size_t)H * E;                      // E*H
    _Float16* fsW1t = gW2t  + (size_t)E * H;                      // NW*H*E
    _Float16* fsW2t = fsW1t + (size_t)NW * H * E;                 // NW*16*H

    auto launchT = [&](const float* in, _Float16* o, int S, int R, int C, int Cp) {
        size_t tot = (size_t)S * Cp * R;
        transpose_f32_to_f16<<<(unsigned)((tot + 255) / 256), 256, 0, stream>>>(in, o, S, R, C, Cp);
    };

    // 1) one-time operand conversions
    launchT(gW1,  gW1t,  1,  E, H,  H);
    launchT(gW2,  gW2t,  1,  H, E,  E);
    launchT(fsW1, fsW1t, NW, E, H,  H);
    launchT(fsW2, fsW2t, NW, H, NL, 16);      // pad 13 -> 16 cols with zeros
    {
        size_t n8 = (size_t)M * E / 8;
        convert_f32_to_f16<<<(unsigned)((n8 + 255) / 256), 256, 0, stream>>>(data, dataH, n8);
        convert_f32_to_f16<<<(unsigned)((n8 + 255) / 256), 256, 0, stream>>>(V, Vh, n8);
    }

    // wide GEMM: grid = (M/128) * (Nc/64) blocks of 256 thr; LDS = 64*(K+8)*2 B
    auto wideGrid = [](int Mm, int Nc) { return (unsigned)((Mm / 128) * (Nc / 64)); };
    auto ldsBytes = [](int K) { return (size_t)64 * (K + 8) * 2; };

    // 2) hoisted scan weights: W[s] = gelu(data@w1s+b1s)@w2s+b2s
    for (int s = 0; s < NW; ++s) {
        gemm_wmma_lds<true, true><<<wideGrid(M, H), 256, ldsBytes(E), stream>>>(
            dataH, fsW1t + (size_t)s * H * E, fsb1 + (size_t)s * H, H,
            Hbuf, M, E, H);
        gemm_wmma_n16<<<(M / 64 + 7) / 8, 256, 0, stream>>>(
            Hbuf, fsW2t + (size_t)s * 16 * H, fsb2 + (size_t)s * NL, NL,
            Wall + (size_t)s * M * 16, M, H);
    }

    // 3) initial MLP on V -> d_out
    gemm_wmma_lds<true, true><<<wideGrid(M, H), 256, ldsBytes(E), stream>>>(
        Vh, gW1t, gb1, H, Hbuf, M, E, H);
    gemm_wmma_lds<false, false><<<wideGrid(M, E), 256, ldsBytes(H), stream>>>(
        Hbuf, gW2t, gb2, E, out, M, H, E);

    // 4) 12 sequential chord-mix steps, ping-pong; 12 is even -> ends in d_out
    for (int s = 0; s < NW; ++s) {
        const float* vi = (s & 1) ? Vbuf : out;
        float*       vo = (s & 1) ? out  : Vbuf;
        chord_step<<<(Bn * N * 64 + 255) / 256, 256, 0, stream>>>(
            vi, vo, Wall + (size_t)s * M * 16, cols);
    }
}
